// MemoryAgent_85040352460962
// MI455X (gfx1250) — matrix operand order repaired and checked by reading, not compiled, not run on hardware
//
#include <hip/hip_runtime.h>

// LSTM forward, MI455X / gfx1250, wave32.
//   K0: f32 -> f16 conversion of x, W_ih, W_hh (one-time, memory bound).
//   K1: xg = x @ W_ih^T + b   (M=32768,N=1024,K=512) pure-f16 WMMA inner loop.
//   K2: persistent 32-wave workgroup scans T=512 steps. h (f16) + c (f32) in LDS;
//       xg_t staged LDS via global_load_async_to_lds_b128 (ASYNCcnt) overlapped
//       with the recurrent WMMA GEMM [64,256]x[256,1024].

typedef _Float16 v16h __attribute__((ext_vector_type(16)));
typedef _Float16 v8h  __attribute__((ext_vector_type(8)));
typedef _Float16 v4h  __attribute__((ext_vector_type(4)));
typedef float    v8f  __attribute__((ext_vector_type(8)));

#define T_DIM 512
#define B_DIM 64
#define I_DIM 512
#define H_DIM 256
#define G_DIM 1024  // 4*H

__device__ __forceinline__ float fsig(float x) {
  return 1.0f / (1.0f + __expf(-x));
}
__device__ __forceinline__ float ftanh(float x) {
  float e = __expf(-2.0f * fabsf(x));
  float t = (1.0f - e) / (1.0f + e);
  return copysignf(t, x);
}

// Build a 16-half A fragment from two contiguous 8-half runs (contiguous k order: element s is k = ka + s).
__device__ __forceinline__ v16h make_a_frag(const _Float16* p /* base + k0 + ka */) {
  v8h lo = *(const v8h*)(p);        // K = ka .. ka+7
  v8h hi = *(const v8h*)(p + 8);    // K = ka+8 .. ka+15
  return __builtin_shufflevector(lo, hi, 0, 1, 2, 3, 4, 5, 6, 7,
                                 8, 9, 10, 11, 12, 13, 14, 15);
}

// Async global->LDS copy of 16 bytes (ASYNCcnt-tracked DMA path).
__device__ __forceinline__ void async_copy_b128(unsigned lds_byte_addr, const void* gaddr) {
  asm volatile("global_load_async_to_lds_b128 %0, %1, off"
               :: "v"(lds_byte_addr), "v"(gaddr)
               : "memory");
}
__device__ __forceinline__ void wait_asynccnt0() {
  asm volatile("s_wait_asynccnt 0x0" ::: "memory");
}

// ---------------------------------------------------------------------------
// K0: elementwise f32 -> f16 (4 elements per thread)
// ---------------------------------------------------------------------------
__global__ __launch_bounds__(256) void cvt_f16_kernel(
    const float* __restrict__ src, _Float16* __restrict__ dst, int n4) {
  int i = blockIdx.x * 256 + threadIdx.x;
  if (i < n4) {
    float4 v = ((const float4*)src)[i];
    v4h o;
    o[0] = (_Float16)v.x;
    o[1] = (_Float16)v.y;
    o[2] = (_Float16)v.z;
    o[3] = (_Float16)v.w;
    ((v4h*)dst)[i] = o;
  }
}

// ---------------------------------------------------------------------------
// K1: xg[m][n] = sum_k x16[m][k] * Wih16[n][k] + b_ih[n] + b_hh[n] (f16 out)
// One wave computes a 16(M) x 64(N) strip; A fragment reused across 4 B tiles.
// ---------------------------------------------------------------------------
__global__ __launch_bounds__(256) void xg_gemm_kernel(
    const _Float16* __restrict__ x16,    // [T*B, I] f16
    const _Float16* __restrict__ Wih16,  // [G, I]  f16
    const float*    __restrict__ bih,
    const float*    __restrict__ bhh,
    _Float16*       __restrict__ xg)     // [T*B, G] f16
{
  const int lane = threadIdx.x & 31;
  const int wid  = blockIdx.x * 8 + (threadIdx.x >> 5);
  const int n_strip = wid & 15;   // 16 strips of 64 cols
  const int m_tile  = wid >> 4;   // 2048 tiles of 16 rows
  const int m0 = m_tile * 16;
  const int n0 = n_strip * 64;

  const int  col = lane & 15;
  const bool hiL = lane >= 16;
  const int  ka  = hiL ? 16 : 0;  // A k base
  const int  kb  = hiL ? 16 : 0;  // B k base

  v8f acc[4] = {};
  const _Float16* xrow = x16 + (size_t)(m0 + col) * I_DIM;

  for (int k0 = 0; k0 < I_DIM; k0 += 32) {
    v16h a = make_a_frag(xrow + k0 + ka);
#pragma unroll
    for (int j = 0; j < 4; ++j) {
      int n = n0 + j * 16 + col;
      // Contiguous k order (same as A): 16 contiguous K per half-wave -> single 32B load.
      v16h b = *(const v16h*)(Wih16 + (size_t)n * I_DIM + k0 + kb);
      acc[j] = __builtin_amdgcn_wmma_f32_16x16x32_f16(
          false, a, false, b, (short)0, acc[j], false, false);
    }
  }

#pragma unroll
  for (int j = 0; j < 4; ++j) {
    int n = n0 + j * 16 + col;
    float bias = bih[n] + bhh[n];
#pragma unroll
    for (int v = 0; v < 8; ++v) {
      int m = m0 + v + (hiL ? 8 : 0);
      xg[(size_t)m * G_DIM + n] = (_Float16)(acc[j][v] + bias);
    }
  }
}

// ---------------------------------------------------------------------------
// K2: sequential scan, single 1024-thread (32-wave) workgroup.
// ---------------------------------------------------------------------------
__global__ __launch_bounds__(1024) void lstm_scan_kernel(
    const _Float16* __restrict__ xg,     // [T*B, G] f16
    const int*      __restrict__ done,   // [T, B]
    const _Float16* __restrict__ Whh16,  // [G, H] f16 (hot in L2: 0.5 MB)
    const float*    __restrict__ h0,
    const float*    __restrict__ c0,
    float*          __restrict__ out)    // [T*B*H] ++ [B*H] ++ [B*H]
{
  __shared__ _Float16 h_s[B_DIM * H_DIM];    // 32 KB
  __shared__ float    c_s[B_DIM * H_DIM];    // 64 KB
  __shared__ _Float16 xg_s[B_DIM * G_DIM];   // 128 KB (async-staged per step)

  const int tid  = threadIdx.x;
  const int lane = tid & 31;
  const int w    = tid >> 5;
  const int  col = lane & 15;
  const bool hiL = lane >= 16;
  const int  ka  = hiL ? 16 : 0;
  const int  kb  = hiL ? 16 : 0;

  const int m_tile = w >> 3;        // 0..3 : batch tile
  const int n_pair = w & 7;         // 0..7 : 32-wide h-column group
  const int m0     = m_tile * 16;
  const int b_row  = m0 + col;      // A-fragment row (batch index)
  const int hbase  = n_pair * 32;

  const unsigned xg_s_base = (unsigned)(uintptr_t)xg_s;

  for (int e = tid; e < B_DIM * H_DIM; e += 1024) {
    h_s[e] = (_Float16)h0[e];
    c_s[e] = c0[e];
  }
  __syncthreads();

  for (int t = 0; t < T_DIM; ++t) {
    // Kick off the async DMA of this step's xg slice (128 KB) into LDS;
    // it overlaps the recurrent WMMA GEMM below.
    {
      const _Float16* src = xg + (size_t)t * B_DIM * G_DIM;
#pragma unroll
      for (int e = 0; e < 8; ++e) {
        int c = e * 1024 + tid;                 // 16-byte chunk index
        async_copy_b128(xg_s_base + (unsigned)c * 16u, src + (size_t)c * 8);
      }
    }

    // Episode reset: h *= (1-done), c *= (1-done)
    for (int e = tid; e < B_DIM * H_DIM; e += 1024) {
      int b = e >> 8;  // H == 256
      if (done[t * B_DIM + b] != 0) {
        h_s[e] = (_Float16)0.0f;
        c_s[e] = 0.0f;
      }
    }
    __syncthreads();

    // g = h @ Whh^T for this wave's (16 batch) x (4 gates x 32 cols) slab.
    v8f acc[8] = {};  // acc[gate*2 + j]
    for (int k0 = 0; k0 < H_DIM; k0 += 32) {
      v16h a = make_a_frag(h_s + b_row * H_DIM + k0 + ka);
#pragma unroll
      for (int q = 0; q < 8; ++q) {
        int gate = q >> 1;
        int n = gate * H_DIM + hbase + (q & 1) * 16 + col;
        v16h bf = *(const v16h*)(Whh16 + (size_t)n * H_DIM + k0 + kb);
        acc[q] = __builtin_amdgcn_wmma_f32_16x16x32_f16(
            false, a, false, bf, (short)0, acc[q], false, false);
      }
    }

    wait_asynccnt0();   // this wave's xg DMA complete
    __syncthreads();    // all waves' DMA + all h_s reads complete

    // Gates + state update; each (b, hcol) owned by exactly one lane slot.
#pragma unroll
    for (int j = 0; j < 2; ++j) {
      int hcol = hbase + j * 16 + col;
#pragma unroll
      for (int v = 0; v < 8; ++v) {
        int b = m0 + v + (hiL ? 8 : 0);
        int base = b * G_DIM;
        float gi = acc[0 + j][v] + (float)xg_s[base + 0 * H_DIM + hcol];
        float gf = acc[2 + j][v] + (float)xg_s[base + 1 * H_DIM + hcol];
        float gg = acc[4 + j][v] + (float)xg_s[base + 2 * H_DIM + hcol];
        float go = acc[6 + j][v] + (float)xg_s[base + 3 * H_DIM + hcol];
        float iv = fsig(gi);
        float fv = fsig(gf);
        float gv = ftanh(gg);
        float ov = fsig(go);
        int sidx = b * H_DIM + hcol;
        float cn = fv * c_s[sidx] + iv * gv;
        float hn = ov * ftanh(cn);
        c_s[sidx] = cn;
        h_s[sidx] = (_Float16)hn;
        out[((size_t)t * B_DIM + b) * H_DIM + hcol] = hn;
        if (t == T_DIM - 1) {  // final h state at f32 precision
          out[(size_t)T_DIM * B_DIM * H_DIM + sidx] = hn;
        }
      }
    }
    __syncthreads();  // h_s/c_s fully written before next step
  }

  // Final c state.
  const size_t off = (size_t)T_DIM * B_DIM * H_DIM + B_DIM * H_DIM;
  for (int e = tid; e < B_DIM * H_DIM; e += 1024) {
    out[off + e] = c_s[e];
  }
}

extern "C" void kernel_launch(void* const* d_in, const int* in_sizes, int n_in,
                              void* d_out, int out_size, void* d_ws, size_t ws_size,
                              hipStream_t stream) {
  const float* x    = (const float*)d_in[0];
  const int*   done = (const int*)d_in[1];
  const float* Wih  = (const float*)d_in[2];
  const float* Whh  = (const float*)d_in[3];
  const float* bih  = (const float*)d_in[4];
  const float* bhh  = (const float*)d_in[5];
  const float* h0   = (const float*)d_in[6];
  const float* c0   = (const float*)d_in[7];
  float* out = (float*)d_out;

  // Workspace layout (f16): xg | x16 | Wih16 | Whh16
  _Float16* xg    = (_Float16*)d_ws;                       // 33,554,432 halves
  _Float16* x16   = xg    + (size_t)T_DIM * B_DIM * G_DIM; // 16,777,216
  _Float16* Wih16 = x16   + (size_t)T_DIM * B_DIM * I_DIM; //    524,288
  _Float16* Whh16 = Wih16 + (size_t)G_DIM * I_DIM;         //    262,144

  const int nx4   = T_DIM * B_DIM * I_DIM / 4;
  const int nwih4 = G_DIM * I_DIM / 4;
  const int nwhh4 = G_DIM * H_DIM / 4;
  cvt_f16_kernel<<<(nx4 + 255) / 256, 256, 0, stream>>>(x, x16, nx4);
  cvt_f16_kernel<<<(nwih4 + 255) / 256, 256, 0, stream>>>(Wih, Wih16, nwih4);
  cvt_f16_kernel<<<(nwhh4 + 255) / 256, 256, 0, stream>>>(Whh, Whh16, nwhh4);

  // 32768 wave-tiles (16x64 each), 8 waves per block -> 4096 blocks.
  xg_gemm_kernel<<<4096, 256, 0, stream>>>(x16, Wih16, bih, bhh, xg);
  // Sequential recurrence: one persistent 32-wave workgroup.
  lstm_scan_kernel<<<1, 1024, 0, stream>>>(xg, done, Whh16, h0, c0, out);
}
